// FFCNET_39513699123391
// MI455X (gfx1250) — matrix operand order repaired and checked by reading, not compiled, not run on hardware
//
#include <hip/hip_runtime.h>
#include <math.h>

#define HP 256
#define HWP 65536
#define NB 4

typedef __attribute__((ext_vector_type(16))) _Float16 v16h;
typedef __attribute__((ext_vector_type(8)))  float    v8f;

__device__ __forceinline__ int imin(int a, int b){ return a < b ? a : b; }
__device__ __forceinline__ int imax(int a, int b){ return a > b ? a : b; }

// ---------------------------------------------------------------- gaussians
__global__ void k_gauss_init(float* gw) {
  int t = threadIdx.x;
  const int   ksA[3]  = {5, 27, 41};
  const float sgA[3]  = {1.5f, 2.0f, 2.8f};
  const int   offA[3] = {0, 8, 40};
  if (t < 3) {
    int ks = ksA[t]; float sg = sgA[t]; float* o = gw + offA[t];
    float sum = 0.f;
    for (int i = 0; i < ks; ++i) {
      float x = i - (ks - 1) * 0.5f;
      float v = expf(-(x * x) / (2.f * sg * sg));
      o[i] = v; sum += v;
    }
    for (int i = 0; i < ks; ++i) o[i] /= sum;
  }
}

// ---------------------------------------------------------------- generic conv
__global__ void k_conv2d(const float* __restrict__ in, const float* __restrict__ w,
                         float* __restrict__ out, const float* __restrict__ resid,
                         int CinTot, int Cin, int cOff, int Cout,
                         int H, int W, int ks, int act)
{
  int idx = blockIdx.x * blockDim.x + threadIdx.x;
  int total = NB * Cout * H * W;
  if (idx >= total) return;
  int x = idx % W; int t = idx / W;
  int y = t % H;  t /= H;
  int co = t % Cout; int b = t / Cout;
  int p = ks >> 1;
  float acc = 0.f;
  const float* wb = w + (size_t)co * Cin * ks * ks;
  for (int ci = 0; ci < Cin; ++ci) {
    const float* ip = in + (size_t)(b * CinTot + cOff + ci) * H * W;
    const float* wc = wb + ci * ks * ks;
    for (int ky = 0; ky < ks; ++ky) {
      int yy = y + ky - p; if (yy < 0 || yy >= H) continue;
      for (int kx = 0; kx < ks; ++kx) {
        int xx = x + kx - p; if (xx < 0 || xx >= W) continue;
        acc += ip[yy * W + xx] * wc[ky * ks + kx];
      }
    }
  }
  if (act == 1) acc = fmaxf(acc, 0.f);
  else if (act == 2) { // tanh-approx gelu
    float x3 = acc * acc * acc;
    acc = 0.5f * acc * (1.f + tanhf(0.7978845608f * (acc + 0.044715f * x3)));
  }
  if (resid) acc += resid[idx];
  out[idx] = acc;
}

// ---------------------------------------------------------------- pixel shuffle r=4
__global__ void k_pixel_shuffle(const float* __restrict__ in, float* __restrict__ out) {
  int idx = blockIdx.x * blockDim.x + threadIdx.x;
  if (idx >= NB * 4 * HWP) return;
  int X = idx % HP; int t = idx / HP;
  int Y = t % HP;  t /= HP;
  int c = t % 4;   int b = t / 4;
  int h = Y >> 2, i2 = Y & 3, w = X >> 2, j2 = X & 3;
  out[idx] = in[(((size_t)b * 64 + c * 16 + i2 * 4 + j2) * 64 + h) * 64 + w];
}

// ---------------------------------------------------------------- reflect blur (sep.)
__global__ void k_blur(const float* __restrict__ in, float* __restrict__ out,
                       const float* __restrict__ gw, int CinTot, int cOff,
                       int ks, int vert)
{
  int idx = blockIdx.x * blockDim.x + threadIdx.x;
  if (idx >= NB * HWP) return;
  int x = idx % HP; int t = idx / HP;
  int y = t % HP;  int b = t / HP;
  const float* ip = in + (size_t)(b * CinTot + cOff) * HWP;
  int p = ks >> 1;
  float acc = 0.f;
  for (int k = 0; k < ks; ++k) {
    int u = (vert ? y : x) + k - p;
    if (u < 0) u = -u;
    if (u >= HP) u = 2 * HP - 2 - u;
    acc += gw[k] * (vert ? ip[u * HP + x] : ip[y * HP + u]);
  }
  out[idx] = acc;
}

// feat = concat([pan-blurP (1ch), lrms0-blurL (1ch), zeros (3ch)])
__global__ void k_featbuild(const float* __restrict__ pan, const float* __restrict__ blurP,
                            const float* __restrict__ lrup, const float* __restrict__ blurL,
                            float* __restrict__ feat)
{
  int idx = blockIdx.x * blockDim.x + threadIdx.x;
  if (idx >= NB * 5 * HWP) return;
  int pix = idx % HWP; int t = idx / HWP;
  int c = t % 5; int b = t / 5;
  float v = 0.f;
  if (c == 0)      v = pan[(size_t)b * HWP + pix] - blurP[(size_t)b * HWP + pix];
  else if (c == 1) v = lrup[(size_t)b * 4 * HWP + pix] - blurL[(size_t)b * HWP + pix];
  feat[idx] = v;
}

// ---------------------------------------------------------------- refine helpers
__global__ void k_mean_hw(const float* __restrict__ in, float* __restrict__ out) {
  __shared__ float sh[256];
  int bc = blockIdx.x;
  const float* p = in + (size_t)bc * HWP;
  float s = 0.f;
  for (int t = threadIdx.x; t < HWP; t += 256) s += p[t];
  sh[threadIdx.x] = s; __syncthreads();
  for (int o = 128; o > 0; o >>= 1) {
    if ((int)threadIdx.x < o) sh[threadIdx.x] += sh[threadIdx.x + o];
    __syncthreads();
  }
  if (threadIdx.x == 0) out[bc] = sh[0] * (1.f / (float)HWP);
}

__global__ void k_ca(const float* __restrict__ mean, const float* __restrict__ w1,
                     const float* __restrict__ w2, float* __restrict__ ca) {
  int t = threadIdx.x; if (t >= NB * 16) return;
  int b = t >> 4, c = t & 15;
  float h[4];
  for (int j = 0; j < 4; ++j) {
    float s = 0.f;
    for (int i2 = 0; i2 < 16; ++i2) s += w1[j * 16 + i2] * mean[b * 16 + i2];
    h[j] = fmaxf(s, 0.f);
  }
  float s = 0.f;
  for (int j = 0; j < 4; ++j) s += w2[c * 4 + j] * h[j];
  ca[t] = 1.f / (1.f + expf(-s));
}

__global__ void k_scale_add(const float* __restrict__ y, const float* __restrict__ bdy,
                            const float* __restrict__ ca, float* __restrict__ out) {
  int idx = blockIdx.x * blockDim.x + threadIdx.x;
  if (idx >= NB * 16 * HWP) return;
  int bc = idx / HWP;
  out[idx] = y[idx] + bdy[idx] * ca[bc];
}

// RIN: ch<8 = l2g+g2g ; ch>=8 = l2l+g2l
__global__ void k_mix(const float* __restrict__ l2l, const float* __restrict__ l2g,
                      const float* __restrict__ g2l, const float* __restrict__ g2g,
                      float* __restrict__ out) {
  int idx = blockIdx.x * blockDim.x + threadIdx.x;
  if (idx >= NB * 16 * HWP) return;
  int pix = idx % HWP; int t = idx / HWP;
  int c = t % 16; int b = t / 16;
  size_t o8 = ((size_t)b * 8 + (c & 7)) * HWP + pix;
  out[idx] = (c < 8) ? (l2g[o8] + g2g[o8]) : (l2l[o8] + g2l[o8]);
}

__global__ void k_copy_ch(const float* __restrict__ src, float* __restrict__ dst,
                          int CtotDst, int chOff) {
  int idx = blockIdx.x * blockDim.x + threadIdx.x;
  if (idx >= NB * 16 * HWP) return;
  int pix = idx % HWP; int t = idx / HWP;
  int c = t % 16; int b = t / 16;
  dst[((size_t)(b * CtotDst + chOff + c)) * HWP + pix] = src[idx];
}

__global__ void k_zero(float* p, int n) {
  int i = blockIdx.x * blockDim.x + threadIdx.x;
  if (i < n) p[i] = 0.f;
}

// ---------------------------------------------------------------- DFT kernels (ortho)
// spectra layout: (B,16,256,129) float, channels 0..7 real, 8..15 imag
__global__ void k_rdft_w(const float* __restrict__ in, float* __restrict__ F,
                         int CinTot, int cOff) {
  __shared__ float cs[256], sn[256];
  for (int t = threadIdx.x; t < 256; t += blockDim.x) {
    float a = 6.283185307f * (float)t * (1.f / 256.f);
    cs[t] = __cosf(a); sn[t] = __sinf(a);
  }
  __syncthreads();
  int idx = blockIdx.x * blockDim.x + threadIdx.x;
  if (idx >= NB * 8 * HP * 129) return;
  int kw = idx % 129; int t = idx / 129;
  int h = t % HP; t /= HP;
  int c = t & 7; int b = t >> 3;
  const float* ip = in + ((size_t)(b * CinTot + cOff + c) * HP + h) * HP;
  float re = 0.f, im = 0.f;
  for (int x = 0; x < HP; ++x) {
    int tt = (kw * x) & 255;
    re += ip[x] * cs[tt];
    im -= ip[x] * sn[tt];   // e^{-i theta}
  }
  F[((size_t)(b * 16 + c)     * HP + h) * 129 + kw] = re * 0.0625f;
  F[((size_t)(b * 16 + c + 8) * HP + h) * 129 + kw] = im * 0.0625f;
}

// complex DFT along H; sgn=-1 forward, +1 inverse
__global__ void k_cdft_h(const float* __restrict__ F, float* __restrict__ G, float sgn) {
  __shared__ float cs[256], sn[256];
  for (int t = threadIdx.x; t < 256; t += blockDim.x) {
    float a = 6.283185307f * (float)t * (1.f / 256.f);
    cs[t] = __cosf(a); sn[t] = __sinf(a);
  }
  __syncthreads();
  int idx = blockIdx.x * blockDim.x + threadIdx.x;
  if (idx >= NB * 8 * HP * 129) return;
  int kw = idx % 129; int t = idx / 129;
  int kh = t % HP; t /= HP;
  int c = t & 7; int b = t >> 3;
  const float* Fr = F + ((size_t)(b * 16 + c)     * HP) * 129 + kw;
  const float* Fi = F + ((size_t)(b * 16 + c + 8) * HP) * 129 + kw;
  float re = 0.f, im = 0.f;
  for (int h = 0; h < HP; ++h) {
    int tt = (kh * h) & 255;
    float fr = Fr[(size_t)h * 129], fi = Fi[(size_t)h * 129];
    re += fr * cs[tt] - sgn * fi * sn[tt];
    im += sgn * fr * sn[tt] + fi * cs[tt];
  }
  G[((size_t)(b * 16 + c)     * HP + kh) * 129 + kw] = re * 0.0625f;
  G[((size_t)(b * 16 + c + 8) * HP + kh) * 129 + kw] = im * 0.0625f;
}

// inverse real DFT along W with Hermitian fold
__global__ void k_irdft_w(const float* __restrict__ Z, float* __restrict__ out) {
  __shared__ float cs[256], sn[256];
  for (int t = threadIdx.x; t < 256; t += blockDim.x) {
    float a = 6.283185307f * (float)t * (1.f / 256.f);
    cs[t] = __cosf(a); sn[t] = __sinf(a);
  }
  __syncthreads();
  int idx = blockIdx.x * blockDim.x + threadIdx.x;
  if (idx >= NB * 8 * HWP) return;
  int x = idx % HP; int t = idx / HP;
  int h = t % HP; t /= HP;
  int c = t & 7; int b = t >> 3;
  const float* Zr = Z + ((size_t)(b * 16 + c)     * HP + h) * 129;
  const float* Zi = Z + ((size_t)(b * 16 + c + 8) * HP + h) * 129;
  float acc = 0.f;
  for (int k = 0; k <= 128; ++k) {
    float f = (k == 0 || k == 128) ? 1.f : 2.f;
    int tt = (k * x) & 255;
    acc += f * (Zr[k] * cs[tt] - Zi[k] * sn[tt]);
  }
  out[((size_t)(b * 8 + c) * HP + h) * HP + x] = acc * 0.0625f;
}

// ---------------------------------------------------------------- MLP weight swizzle
// dst layout: [kt][nt][lane][e] with k = kt*32 + e + 16*(lane>>4), n = nt*16 + (lane&15)
__global__ void k_prep_w(const float* __restrict__ W, _Float16* __restrict__ dst,
                         int Ksrc, int Kpad, int N) {
  int idx = blockIdx.x * blockDim.x + threadIdx.x;
  if (idx >= Kpad * N) return;
  int e = idx & 15; int lane = (idx >> 4) & 31; int tile = idx >> 9;
  int NT = N >> 4;
  int nt = tile % NT; int kt = tile / NT;
  int k = kt * 32 + e + 16 * (lane >> 4);
  int n = nt * 16 + (lane & 15);
  float v = (k < Ksrc) ? W[(size_t)k * N + n] : 0.f;
  dst[idx] = (_Float16)v;
}

// ---------------------------------------------------------------- WMMA GEMM
// A: [M,K] f16 row-major.  Bw: swizzled f16 weights.  out = act(A@W + bias)
// 8 waves/block; each wave computes 16 rows x (16*CT) cols: the A fragment is
// loaded once per K-step and reused by CT WMMAs (4x less A traffic, denser WMMA).
// grid = (N/(16*CT), M/128)
template <int CT>
__global__ __launch_bounds__(256)
void k_gemm(const _Float16* __restrict__ A, const _Float16* __restrict__ Bw,
            const float* __restrict__ bias, float* __restrict__ outF32,
            _Float16* __restrict__ outF16, int K, int N, int relu)
{
  int lane = threadIdx.x & 31, wave = threadIdx.x >> 5;
  int ntBase = blockIdx.x * CT;
  int mBase = blockIdx.y * 128 + wave * 16;
  int grp = lane >> 4;
  int NT = N >> 4;
  const v8f vzero = {0.f, 0.f, 0.f, 0.f, 0.f, 0.f, 0.f, 0.f};
  v8f acc[CT];
#pragma unroll
  for (int q = 0; q < CT; ++q) acc[q] = vzero;

  const _Float16* arow = A + (size_t)(mBase + (lane & 15)) * K;
  for (int kt = 0; kt < K; kt += 32) {
    union { v16h v; unsigned u[8]; } a;
    const _Float16* ap = arow + kt + 16 * grp;
#pragma unroll
    for (int m = 0; m < 8; ++m) {
      int koff = 2 * m;  // K contiguous 0..15 (+16 per grp)
      a.u[m] = *(const unsigned*)(ap + koff);
    }
#pragma unroll
    for (int q = 0; q < CT; ++q) {
      union { v16h v; unsigned u[8]; } b;
      const unsigned* bp = (const unsigned*)
          (Bw + ((((size_t)(kt >> 5)) * NT + (ntBase + q)) * 32 + lane) * 16);
#pragma unroll
      for (int m = 0; m < 8; ++m) b.u[m] = bp[m];
      acc[q] = __builtin_amdgcn_wmma_f32_16x16x32_f16(false, a.v, false, b.v,
                                                      (short)0, acc[q], false, false);
    }
  }
  int r0 = mBase + 8 * grp;
#pragma unroll
  for (int q = 0; q < CT; ++q) {
    int col = (ntBase + q) * 16 + (lane & 15);
    float bv = bias[col];
#pragma unroll
    for (int v = 0; v < 8; ++v) {
      float val = acc[q][v] + bv;
      if (relu) val = fmaxf(val, 0.f);
      size_t o = (size_t)(r0 + v) * N + col;
      if (outF32) outF32[o] = val;
      else        outF16[o] = (_Float16)val;
    }
  }
}

// ---------------------------------------------------------------- LIIF
__device__ __forceinline__ void liif_geom(int i, int j, float vx, float vy,
                                          int& i0, int& i1, float& rel0, float& rel1) {
  const float eps = 1e-6f;
  float c0 = -1.f + (2 * i + 1) * (1.f / 256.f);
  float c1 = -1.f + (2 * j + 1) * (1.f / 256.f);
  float a0 = fminf(fmaxf(c0 + vx * (1.f / 256.f) + eps, -1.f + eps), 1.f - eps);
  float a1 = fminf(fmaxf(c1 + vy * (1.f / 256.f) + eps, -1.f + eps), 1.f - eps);
  i0 = imin(imax((int)floorf((a0 + 1.f) * 128.f), 0), 255);
  i1 = imin(imax((int)floorf((a1 + 1.f) * 128.f), 0), 255);
  rel0 = (c0 - (-1.f + (2 * i0 + 1) * (1.f / 256.f))) * 256.f;
  rel1 = (c1 - (-1.f + (2 * i1 + 1) * (1.f / 256.f))) * 256.f;
}

// build one (corner, batch) chunk of MLP inputs: rows [HW, 160] f16
__global__ void k_liif_build(const float* __restrict__ feat, _Float16* __restrict__ Arow,
                             int b, float vx, float vy) {
  int idx = blockIdx.x * blockDim.x + threadIdx.x;
  if (idx >= HWP * 160) return;
  int e = idx % 160; int p = idx / 160;
  int i = p >> 8, j = p & 255;
  int i0, i1; float rel0, rel1;
  liif_geom(i, j, vx, vy, i0, i1, rel0, rel1);
  float v = 0.f;
  if (e < 144) {
    int bi = e >> 4, c = e & 15;
    int di = bi / 3, dj = bi % 3;
    int yy = i0 + di - 1, xx = i1 + dj - 1;
    if (yy >= 0 && yy < 256 && xx >= 0 && xx < 256)
      v = feat[(((size_t)b * 16 + c) * 256 + yy) * 256 + xx];
  } else if (e == 144) v = rel0;
  else if (e == 145)   v = rel1;
  else if (e <= 147)   v = 2.f;   // cell * h, cell * w
  Arow[idx] = (_Float16)v;
}

__global__ void k_liif_combine(const float* __restrict__ pred, float* __restrict__ featO,
                               int b, int corner) {
  int idx = blockIdx.x * blockDim.x + threadIdx.x;
  if (idx >= HWP * 16) return;
  int c = idx & 15; int p = idx >> 4;
  int i = p >> 8, j = p & 255;
  float areas[4];
  for (int cr = 0; cr < 4; ++cr) {
    float vx = (cr < 2) ? -1.f : 1.f;
    float vy = (cr & 1) ? 1.f : -1.f;
    int i0, i1; float r0, r1;
    liif_geom(i, j, vx, vy, i0, i1, r0, r1);
    areas[cr] = fabsf(r0 * r1) + 1e-9f;
  }
  float tot = areas[0] + areas[1] + areas[2] + areas[3];
  float w = areas[3 - corner] / tot;   // swap (0,3) and (1,2)
  featO[(((size_t)b * 16 + c) * 256 + i) * 256 + j] += pred[(size_t)p * 16 + c] * w;
}

// ================================================================ host
extern "C" void kernel_launch(void* const* d_in, const int* in_sizes, int n_in,
                              void* d_out, int out_size, void* d_ws, size_t ws_size,
                              hipStream_t stream) {
  (void)in_sizes; (void)n_in; (void)out_size; (void)ws_size;
  const float* lrms      = (const float*)d_in[0];
  const float* pan       = (const float*)d_in[1];
  const float* w_conv_ps = (const float*)d_in[2];
  const float* hor_w1    = (const float*)d_in[3];
  const float* hor_w2    = (const float*)d_in[4];
  const float* ffc_l2l   = (const float*)d_in[5];
  const float* ffc_l2g   = (const float*)d_in[6];
  const float* ffc_g2l   = (const float*)d_in[7];
  const float* ffc_spec  = (const float*)d_in[8];
  const float* ref_in    = (const float*)d_in[9];
  const float* ref_b1    = (const float*)d_in[10];
  const float* ref_b2    = (const float*)d_in[11];
  const float* ref_ca1   = (const float*)d_in[12];
  const float* ref_ca2   = (const float*)d_in[13];
  const float* ref_out   = (const float*)d_in[14];
  const float* w_fuse    = (const float*)d_in[15];
  const float* w_liif    = (const float*)d_in[16];
  const float* w_hp      = (const float*)d_in[17];
  const float* mlp_w_in  = (const float*)d_in[18];
  const float* mlp_b_in  = (const float*)d_in[19];
  const float* mlp_w_h   = (const float*)d_in[20];
  const float* mlp_b_h   = (const float*)d_in[21];
  const float* mlp_w_out = (const float*)d_in[22];
  const float* mlp_b_out = (const float*)d_in[23];
  float* outp = (float*)d_out;

  char* base = (char*)d_ws;
  size_t off = 0;
  auto alloc = [&](size_t bytes) -> char* {
    char* r = base + off;
    off += bytes; off = (off + 255) & ~(size_t)255;
    return r;
  };

  const size_t C16 = (size_t)NB * 16 * HWP;
  const size_t C8b = (size_t)NB * 8 * HWP * 4;
  const size_t FQn = (size_t)NB * 16 * HP * 129;

  float* GW    = (float*)alloc(128 * 4);
  float* PSOUT = (float*)alloc((size_t)NB * 64 * 64 * 64 * 4);
  float* LRUP  = (float*)alloc((size_t)NB * 4 * HWP * 4);
  float* TB1   = (float*)alloc((size_t)NB * HWP * 4);
  float* TB2   = (float*)alloc((size_t)NB * HWP * 4);
  float* TB3   = (float*)alloc((size_t)NB * HWP * 4);
  float* X0    = (float*)alloc(C16 * 4);
  float* X1    = (float*)alloc(C16 * 4);
  float* Yb    = (float*)alloc(C16 * 4);
  float* T1    = (float*)alloc(C16 * 4);
  float* BDY   = (float*)alloc(C16 * 4);
  float* T2    = (float*)alloc(C16 * 4);
  float* RIN   = (float*)alloc(C16 * 4);
  float* F0    = (float*)alloc(C16 * 4);
  float* F1    = (float*)alloc(C16 * 4);
  float* F2    = (float*)alloc(C16 * 4);
  float* FU0   = (float*)alloc(C16 * 4);
  float* FU1   = (float*)alloc(C16 * 4);
  float* FU2   = (float*)alloc(C16 * 4);
  float* FEATA = (float*)alloc(C16 * 4);
  float* FEATO = (float*)alloc(C16 * 4);
  float* L2L   = (float*)alloc(C8b);
  float* L2G   = (float*)alloc(C8b);
  float* G2L   = (float*)alloc(C8b);
  float* G2G   = (float*)alloc(C8b);
  float* FQ1   = (float*)alloc(FQn * 4);
  float* FQ2   = (float*)alloc(FQn * 4);
  float* FQ3   = (float*)alloc(FQn * 4);
  float* CAT48 = (float*)alloc((size_t)NB * 48 * HWP * 4);
  float* MEAN  = (float*)alloc(64 * 4);
  float* CAb   = (float*)alloc(64 * 4);
  _Float16* WSI = (_Float16*)alloc((size_t)160 * 256 * 2);
  _Float16* WSH = (_Float16*)alloc((size_t)3 * 256 * 256 * 2);
  _Float16* WSO = (_Float16*)alloc((size_t)256 * 16 * 2);
  _Float16* AIN = (_Float16*)alloc((size_t)HWP * 160 * 2);
  _Float16* Hb1 = (_Float16*)alloc((size_t)HWP * 256 * 2);
  _Float16* Hb2 = (_Float16*)alloc((size_t)HWP * 256 * 2);
  float* PRED   = (float*)alloc((size_t)HWP * 16 * 4);

#define GRID(n) dim3(((n) + 255) / 256), dim3(256), 0, stream

  auto conv = [&](const float* in, const float* w, float* out, const float* resid,
                  int CinTot, int Cin, int cOff, int Cout, int H, int W, int ks, int act) {
    int total = NB * Cout * H * W;
    k_conv2d<<<GRID(total)>>>(in, w, out, resid, CinTot, Cin, cOff, Cout, H, W, ks, act);
  };

  auto refine = [&](const float* x, int i, float* dst) {
    const float* win  = ref_in  + (size_t)i * 2304;
    const float* wb1  = ref_b1  + (size_t)i * 2304;
    const float* wb2  = ref_b2  + (size_t)i * 2304;
    const float* wca1 = ref_ca1 + (size_t)i * 64;
    const float* wca2 = ref_ca2 + (size_t)i * 64;
    const float* wout = ref_out + (size_t)i * 2304;
    conv(x,  win, Yb,  nullptr, 16, 16, 0, 16, HP, HP, 3, 0);
    conv(Yb, wb1, T1,  nullptr, 16, 16, 0, 16, HP, HP, 3, 1);
    conv(T1, wb2, BDY, nullptr, 16, 16, 0, 16, HP, HP, 3, 0);
    k_mean_hw<<<NB * 16, 256, 0, stream>>>(BDY, MEAN);
    k_ca<<<1, 64, 0, stream>>>(MEAN, wca1, wca2, CAb);
    k_scale_add<<<GRID((int)C16)>>>(Yb, BDY, CAb, T2);
    conv(T2, wout, dst, nullptr, 16, 16, 0, 16, HP, HP, 3, 0);
  };

  auto ffc = [&](const float* x, int j) {
    const float* wl2l  = ffc_l2l  + (size_t)j * 576;
    const float* wl2g  = ffc_l2g  + (size_t)j * 576;
    const float* wg2l  = ffc_g2l  + (size_t)j * 576;
    const float* wspec = ffc_spec + (size_t)j * 256;
    conv(x, wl2l, L2L, nullptr, 16, 8, 0, 8, HP, HP, 3, 0);
    conv(x, wl2g, L2G, nullptr, 16, 8, 0, 8, HP, HP, 3, 0);
    conv(x, wg2l, G2L, nullptr, 16, 8, 8, 8, HP, HP, 3, 0);
    int nf = NB * 8 * HP * 129;
    k_rdft_w<<<GRID(nf)>>>(x, FQ1, 16, 8);
    k_cdft_h<<<GRID(nf)>>>(FQ1, FQ2, -1.f);
    conv(FQ2, wspec, FQ3, nullptr, 16, 16, 0, 16, HP, 129, 1, 1);
    k_cdft_h<<<GRID(nf)>>>(FQ3, FQ1, 1.f);
    k_irdft_w<<<GRID(NB * 8 * HWP)>>>(FQ1, G2G);
  };

  // ---- stem
  k_gauss_init<<<1, 64, 0, stream>>>(GW);
  conv(lrms, w_conv_ps, PSOUT, nullptr, 4, 4, 0, 64, 64, 64, 3, 0);
  k_pixel_shuffle<<<GRID(NB * 4 * HWP)>>>(PSOUT, LRUP);

  const int ksA[3] = {5, 27, 41};
  const int gwO[3] = {0, 8, 40};
  float* FUSED[3] = {FU0, FU1, FU2};

  for (int br = 0; br < 3; ++br) {
    int ks = ksA[br]; const float* gk = GW + gwO[br];
    int nbl = NB * HWP;
    k_blur<<<GRID(nbl)>>>(pan,  TB1, gk, 1, 0, ks, 1);
    k_blur<<<GRID(nbl)>>>(TB1,  TB2, gk, 1, 0, ks, 0);
    k_blur<<<GRID(nbl)>>>(LRUP, TB1, gk, 4, 0, ks, 1);
    k_blur<<<GRID(nbl)>>>(TB1,  TB3, gk, 1, 0, ks, 0);
    k_featbuild<<<GRID(NB * 5 * HWP)>>>(pan, TB2, LRUP, TB3, X0);
    // hornet
    conv(X0, hor_w1 + (size_t)br * 720,  X1, nullptr, 5, 5, 0, 16, HP, HP, 3, 2);
    conv(X1, hor_w2 + (size_t)br * 2304, X0, nullptr, 16, 16, 0, 16, HP, HP, 3, 0);
    // refine/ffc chain
    refine(X0, 3 * br, F0);
    ffc(F0, 2 * br);
    k_mix<<<GRID((int)C16)>>>(L2L, L2G, G2L, G2G, RIN);
    refine(RIN, 3 * br + 1, F1);
    ffc(F1, 2 * br + 1);
    k_mix<<<GRID((int)C16)>>>(L2L, L2G, G2L, G2G, RIN);
    refine(RIN, 3 * br + 2, F2);
    // fuse branch
    k_copy_ch<<<GRID((int)C16)>>>(F0, CAT48, 48, 0);
    k_copy_ch<<<GRID((int)C16)>>>(F1, CAT48, 48, 16);
    k_copy_ch<<<GRID((int)C16)>>>(F2, CAT48, 48, 32);
    conv(CAT48, w_fuse + (size_t)br * 768, FUSED[br], nullptr, 48, 48, 0, 16, HP, HP, 1, 0);
  }
  k_copy_ch<<<GRID((int)C16)>>>(FU0, CAT48, 48, 0);
  k_copy_ch<<<GRID((int)C16)>>>(FU1, CAT48, 48, 16);
  k_copy_ch<<<GRID((int)C16)>>>(FU2, CAT48, 48, 32);
  conv(CAT48, w_liif, FEATA, nullptr, 48, 48, 0, 16, HP, HP, 1, 0);

  // ---- MLP weight prep (f32 -> f16 WMMA-B swizzle)
  k_prep_w<<<GRID(160 * 256)>>>(mlp_w_in, WSI, 148, 160, 256);
  for (int l = 0; l < 3; ++l)
    k_prep_w<<<GRID(256 * 256)>>>(mlp_w_h + (size_t)l * 65536, WSH + (size_t)l * 65536,
                                  256, 256, 256);
  k_prep_w<<<GRID(256 * 16)>>>(mlp_w_out, WSO, 256, 256, 16);

  k_zero<<<GRID((int)C16)>>>(FEATO, (int)C16);

  // ---- LIIF MLP: per (corner, batch) chunks of M = 65536 rows
  dim3 gHid(256 / 64, HWP / 128);   // CT=4 -> 16x64 per wave
  dim3 gOut(1,        HWP / 128);   // CT=1 for N=16
  for (int cr = 0; cr < 4; ++cr) {
    float vx = (cr < 2) ? -1.f : 1.f;
    float vy = (cr & 1) ? 1.f : -1.f;
    for (int b = 0; b < NB; ++b) {
      k_liif_build<<<GRID(HWP * 160)>>>(FEATA, AIN, b, vx, vy);
      k_gemm<4><<<gHid, 256, 0, stream>>>(AIN, WSI, mlp_b_in, nullptr, Hb1, 160, 256, 1);
      k_gemm<4><<<gHid, 256, 0, stream>>>(Hb1, WSH,          mlp_b_h,       nullptr, Hb2, 256, 256, 1);
      k_gemm<4><<<gHid, 256, 0, stream>>>(Hb2, WSH +  65536, mlp_b_h + 256, nullptr, Hb1, 256, 256, 1);
      k_gemm<4><<<gHid, 256, 0, stream>>>(Hb1, WSH + 131072, mlp_b_h + 512, nullptr, Hb2, 256, 256, 1);
      k_gemm<1><<<gOut, 256, 0, stream>>>(Hb2, WSO, mlp_b_out, PRED, nullptr, 256, 16, 0);
      k_liif_combine<<<GRID(HWP * 16)>>>(PRED, FEATO, b, cr);
    }
  }

  // ---- final: conv(fo, w_hp) + lrms_up
  conv(FEATO, w_hp, outp, LRUP, 16, 16, 0, 4, HP, HP, 3, 0);
#undef GRID
}
